// FSAFTarget_30502857736596
// MI455X (gfx1250) — matrix operand order repaired and checked by reading, not compiled, not run on hardware
//
#include <hip/hip_runtime.h>
#include <stdint.h>

typedef __attribute__((ext_vector_type(16))) _Float16 v16h;
typedef __attribute__((ext_vector_type(8)))  float    v8f;

#define NUM_LEVELS 5
#define MAX_GT     64
#define NUM_CLASSES 80
#define HW_TOTAL   21824      // 128^2+64^2+32^2+16^2+8^2 ; multiple of 16
#define BATCH      16
#define BLOCK      256
#define BIG_AREA   10000000.0f

// Flat output offsets (floats), concatenation order of the reference tuple:
// cls_t [16,21824,80] | cls_m [16,21824] | num_pos [16] | reg_t [16,21824,4] | reg_m [16,21824]
#define CLSM_OFF  27934720LL
#define NP_OFF    28283904LL
#define REGT_OFF  28283920LL
#define REGM_OFF  29680656LL

__global__ __launch_bounds__(BLOCK)
void fsaf_target_kernel(const int* __restrict__ gt_levels,   // [16,64] int32
                        const float* __restrict__ gt_boxes,  // [16,64,5] float (x1,y1,x2,y2,label)
                        float* __restrict__ out)
{
    const int img   = blockIdx.y;
    const int chunk = blockIdx.x;
    const int tid   = threadIdx.x;

    __shared__ float    s_box[MAX_GT][4];
    __shared__ int      s_label[MAX_GT];
    __shared__ int      s_level[MAX_GT];
    __shared__ float    s_area[MAX_GT];
    __shared__ uint32_t s_rect[NUM_LEVELS][MAX_GT][2]; // [0]=pos rect, [1]=ignore rect (byte-packed x1,y1,x2,y2)
    __shared__ uint32_t s_pc[BLOCK];                   // (chosen<<1)|pos per pixel
    __shared__ int      s_np;

    if (tid == 0) s_np = 0;
    if (tid < MAX_GT) {
        const float* g = gt_boxes + ((long long)img * MAX_GT + tid) * 5;
        float x1 = g[0], y1 = g[1], x2 = g[2], y2 = g[3];
        s_box[tid][0] = x1; s_box[tid][1] = y1; s_box[tid][2] = x2; s_box[tid][3] = y2;
        s_label[tid]  = (int)g[4];
        s_level[tid]  = gt_levels[img * MAX_GT + tid];
        s_area[tid]   = (x2 - x1) * (y2 - y1);
    }
    __syncthreads();

    // ---- Precompute per-(level, gt) shrunken rects (POS scale 0.2 -> a=0.4; IGN 0.5 -> a=0.25) ----
    for (int idx = tid; idx < NUM_LEVELS * MAX_GT; idx += BLOCK) {
        int lvl = idx >> 6, n = idx & 63;
        float stride = (float)(8 << lvl);
        float fdim   = (float)(128 >> lvl);
        float inv_s  = 1.0f / stride;
        float px1 = s_box[n][0] * inv_s, py1 = s_box[n][1] * inv_s;
        float px2 = s_box[n][2] * inv_s, py2 = s_box[n][3] * inv_s;
        float w = px2 - px1, h = py2 - py1;
        const float av[2] = {0.4f, 0.25f};
        #pragma unroll
        for (int k = 0; k < 2; k++) {
            float a   = av[k];
            float nx1 = floorf(px1 + a * w);
            float ny1 = floorf(py1 + a * h);
            float nx2 = ceilf (px2 - a * w);
            float ny2 = ceilf (py2 - a * h);
            float cx2 = fminf(fmaxf(fmaxf(nx2, nx1 + 1.0f), 0.0f), fdim);
            float cy2 = fminf(fmaxf(fmaxf(ny2, ny1 + 1.0f), 0.0f), fdim);
            float cx1 = fminf(fmaxf(nx1, 0.0f), fdim - 1.0f);
            float cy1 = fminf(fmaxf(ny1, 0.0f), fdim - 1.0f);
            uint32_t p = (uint32_t)(int)cx1 | ((uint32_t)(int)cy1 << 8) |
                         ((uint32_t)(int)cx2 << 16) | ((uint32_t)(int)cy2 << 24);
            s_rect[lvl][n][k] = p;
        }
    }
    __syncthreads();

    // ---- Per-pixel scan of 64 GT boxes ----
    const int  hw       = chunk * BLOCK + tid;
    const bool in_range = hw < HW_TOTAL;
    const int  hwc      = in_range ? hw : 0;

    int lvl, off;
    if      (hwc < 16384) { lvl = 0; off = 0;     }
    else if (hwc < 20480) { lvl = 1; off = 16384; }
    else if (hwc < 21504) { lvl = 2; off = 20480; }
    else if (hwc < 21760) { lvl = 3; off = 21504; }
    else                  { lvl = 4; off = 21760; }
    const int   fdim   = 128 >> lvl;
    const int   rel    = hwc - off;
    const int   y      = rel >> (7 - lvl);
    const int   x      = rel & (fdim - 1);
    const float stride = (float)(8 << lvl);

    bool  any_pos = false, any_ign = false;
    float best_area = BIG_AREA;
    int   best = 0;
    for (int n = 0; n < MAX_GT; n++) {
        bool match = (s_level[n] == lvl);
        uint32_t pp = s_rect[lvl][n][0];
        uint32_t ip = s_rect[lvl][n][1];
        int px1 = pp & 255, py1 = (pp >> 8) & 255, px2 = (pp >> 16) & 255, py2 = pp >> 24;
        int ix1 = ip & 255, iy1 = (ip >> 8) & 255, ix2 = (ip >> 16) & 255, iy2 = ip >> 24;
        bool ipos = match & (x >= px1) & (x < px2) & (y >= py1) & (y < py2);
        bool iign = match & (x >= ix1) & (x < ix2) & (y >= iy1) & (y < iy2);
        any_pos |= ipos;
        any_ign |= iign;
        float a = ipos ? s_area[n] : BIG_AREA;
        if (a < best_area) { best_area = a; best = n; }   // first-min == jnp.argmin
    }
    if (!in_range) { any_pos = false; any_ign = false; best = 0; }
    const float posf = any_pos ? 1.0f : 0.0f;

    if (in_range) {
        long long base = (long long)img * HW_TOTAL + hw;
        out[CLSM_OFF + base] = (any_pos || !any_ign) ? 1.0f : 0.0f;  // mx==2 or mx==-1
        out[REGM_OFF + base] = posf;
        float cb0 = s_box[best][0], cb1 = s_box[best][1];
        float cb2 = s_box[best][2], cb3 = s_box[best][3];
        float sx = ((float)x + 0.5f) * stride;
        float sy = ((float)y + 0.5f) * stride;
        float4 rt;
        rt.x = (sx - cb0) * 0.25f * posf;
        rt.y = (sy - cb1) * 0.25f * posf;
        rt.z = (cb2 - sx) * 0.25f * posf;
        rt.w = (cb3 - sy) * 0.25f * posf;
        *(float4*)(out + REGT_OFF + base * 4) = rt;
        if (any_pos) atomicAdd(&s_np, 1);
    }
    s_pc[tid] = ((uint32_t)best << 1) | (any_pos ? 1u : 0u);
    __syncthreads();

    if (tid == 0 && s_np > 0)
        atomicAdd(out + NP_OFF + img, (float)s_np);   // integer-valued -> exact & deterministic

    // ---- WMMA one-hot expansion: CLS[16x80] = S[16x64] x O[64x80], K=64 via two chained x32 ----
    const int lane = tid & 31;
    // force wave index scalar so the tile-store guard is a single s_cbranch (no per-store EXEC juggling)
    const int wave = __builtin_amdgcn_readfirstlane(tid >> 5);
    const int hi   = lane >> 4;       // half-wave select
    const int r    = lane & 15;       // row (A/pixel) or col (B/D/class) within tile
    const int waveBase = wave * 32;
    float* clsout = out + (long long)img * HW_TOTAL * NUM_CLASSES;

    for (int ct = 0; ct < 5; ct++) {                 // 5 class tiles of 16 -> 80 classes
        const int cbase = ct * 16;
        // B (32x16 f16): lane holds col N=r; element e -> GT row K = e + 16*hi (+32 for step 1)
        v16h b0, b1;
        #pragma unroll
        for (int e = 0; e < 16; e++) {
            int n0 = e + 16 * hi;
            b0[e] = (s_label[n0]      == cbase + r) ? (_Float16)1.0f : (_Float16)0.0f;
            b1[e] = (s_label[n0 + 32] == cbase + r) ? (_Float16)1.0f : (_Float16)0.0f;
        }
        #pragma unroll
        for (int t = 0; t < 2; t++) {                // two 16-pixel tile rows per wave
            uint32_t pc  = s_pc[waveBase + 16 * t + r];
            bool     pos = pc & 1u;
            int      ch  = (int)(pc >> 1);
            // A (16x32 f16): lane holds row M=r; element e -> K = e + 16*hi
            v16h a0, a1;
            #pragma unroll
            for (int e = 0; e < 16; e++) {
                int K = e + 16 * hi;
                a0[e] = (pos && ch == K)      ? (_Float16)1.0f : (_Float16)0.0f;
                a1[e] = (pos && ch == K + 32) ? (_Float16)1.0f : (_Float16)0.0f;
            }
            v8f c = {};
            c = __builtin_amdgcn_wmma_f32_16x16x32_f16(false, a0, false, b0, (short)0, c, false, false);
            c = __builtin_amdgcn_wmma_f32_16x16x32_f16(false, a1, false, b1, (short)0, c, false, false);

            // D: VGPR j -> row M = j + 8*hi, col N = r.
            // HW_TOTAL % 16 == 0 => each 16-pixel tile row is all-in or all-out:
            // guard is wave-uniform -> single scalar branch, 8 unguarded coalesced stores.
            const int tilebase = chunk * BLOCK + waveBase + 16 * t;
            if (tilebase < HW_TOTAL) {
                float* p = clsout + (long long)(tilebase + 8 * hi) * NUM_CLASSES + cbase + r;
                #pragma unroll
                for (int j = 0; j < 8; j++)
                    p[j * NUM_CLASSES] = c[j];
            }
        }
    }
}

extern "C" void kernel_launch(void* const* d_in, const int* in_sizes, int n_in,
                              void* d_out, int out_size, void* d_ws, size_t ws_size,
                              hipStream_t stream) {
    const int*   gt_levels = (const int*)d_in[0];    // batch_gt_box_levels [16,64]
    const float* gt_boxes  = (const float*)d_in[1];  // batch_gt_boxes [16,64,5]
    // d_in[2] (feature_shapes) is a compile-time constant pyramid; unused.
    float* out = (float*)d_out;

    // num_pos accumulators must start at zero (graph-capturable memset node)
    hipMemsetAsync(out + NP_OFF, 0, BATCH * sizeof(float), stream);

    dim3 grid((HW_TOTAL + BLOCK - 1) / BLOCK, BATCH);  // 86 x 16 blocks
    fsaf_target_kernel<<<grid, BLOCK, 0, stream>>>(gt_levels, gt_boxes, out);
}
